// FCOSModule_18339510354705
// MI455X (gfx1250) — matrix operand order repaired and checked by reading, not compiled, not run on hardware
//
#include <hip/hip_runtime.h>
#include <math.h>

typedef __attribute__((ext_vector_type(16))) _Float16 v16h;
typedef __attribute__((ext_vector_type(8)))  _Float16 v8h;
typedef __attribute__((ext_vector_type(8)))  float    v8f;

#define FCOS_EPS 1e-5f

__device__ inline v8h v8h_zero() {
    v8h z;
    for (int i = 0; i < 8; ++i) z[i] = (_Float16)0.f;
    return z;
}

__device__ inline v16h cat16(v8h lo, v8h hi) {
    return __builtin_shufflevector(lo, hi, 0,1,2,3,4,5,6,7,8,9,10,11,12,13,14,15);
}

// CDNA5 async global->LDS copy (bypasses VGPRs, tracked by ASYNCcnt).
__device__ __forceinline__ void async_ld_b128(unsigned lds_byte_off, unsigned long long ga) {
    asm volatile("global_load_async_to_lds_b128 %0, %1, off"
                 :: "v"(lds_byte_off), "v"(ga)
                 : "memory");
}
__device__ __forceinline__ void s_wait_async0() {
    asm volatile("s_wait_asynccnt 0" ::: "memory");
}

// ---------------------------------------------------------------------------
// Implicit-GEMM 3x3 conv. IC=256. Input is zero-halo-padded f16 NHWC:
//   in[((b*HW2 + (y+ky)*W2 + (x+kx)) * 256 + ic]   (halo => no bounds checks)
// Weights f16 [k9][OCpad][256]. Workgroup: 256 threads = 8 waves.
// Tile: 128 spatial (M) x 64 out-channels (N); wave tile 32x32.
// 36 K-stages of 64; per stage: 6 async global->LDS issues, 16 ds_load_b128,
// 8 v_wmma_f32_16x16x32_f16, one barrier. Double-buffered LDS (55.3 KB).
// Global addresses maintained incrementally (no div/mul in the hot loop).
// mode 0: f16 padded-NHWC out (bias added)
// mode 1: f32 scatter into concat output layout (logits / ctr)
// mode 2: like 1 but out = exp(scale * x)  (reg branch)
// ---------------------------------------------------------------------------
__global__ __launch_bounds__(256) void conv3x3_wmma(
    const _Float16* __restrict__ in, const _Float16* __restrict__ wt,
    const float* __restrict__ bias, _Float16* __restrict__ out16,
    float* __restrict__ out32, const float* __restrict__ scale,
    int Bsz, int W, int HW, int W2, int HW2, int OC, int OCpad, int mode,
    long long out_off, long long TOT, int ocbase)
{
    __shared__ _Float16 As[2 * 128 * 72];
    __shared__ _Float16 Bs[2 * 64 * 72];

    const int t    = threadIdx.x;
    const int lane = t & 31;
    const int wv   = t >> 5;
    const int M    = Bsz * HW;
    const int m0   = blockIdx.x * 128;
    const int n0   = blockIdx.y * 64;

    // ---- staging decomposition: rows x 4 segments of 8 halves (16B) ----
    const int sp  = t >> 2;          // 0..63 (A rows sp and sp+64)
    const int seg = (t & 3) * 8;
    unsigned long long ga0, ga1, gw;
    {
        int mg = m0 + sp;      if (mg >= M) mg = 0;
        int b = mg / HW; int p = mg - b * HW; int y = p / W; int x = p - y * W;
        ga0 = (unsigned long long)(uintptr_t)(
            in + ((long long)(b * HW2 + y * W2 + x)) * 256 + seg);
        mg = m0 + sp + 64;     if (mg >= M) mg = 0;
        b = mg / HW; p = mg - b * HW; y = p / W; x = p - y * W;
        ga1 = (unsigned long long)(uintptr_t)(
            in + ((long long)(b * HW2 + y * W2 + x)) * 256 + seg);
        gw = (unsigned long long)(uintptr_t)(
            wt + ((long long)(n0 + sp)) * 256 + seg);
    }
    const long long aD = (long long)(W2 - 3) * 512;   // extra bytes on tap-row step
    const long long wD = (long long)(OCpad - 1) * 512; // extra bytes per k9 (weights)
    int ktc = 0, k3 = 0;

    const unsigned asBase = (unsigned)(uintptr_t)(&As[0]);
    const unsigned bsBase = (unsigned)(uintptr_t)(&Bs[0]);
    const unsigned lA0 = asBase + (unsigned)(sp * 72 + seg) * 2;
    const unsigned lA1 = asBase + (unsigned)((sp + 64) * 72 + seg) * 2;
    const unsigned lB  = bsBase + (unsigned)(sp * 72 + seg) * 2;

    // issue one 64-K stage into LDS buffer `buf`, then advance addresses
    auto issue = [&](int buf) {
        const unsigned oa = (unsigned)(buf * 128 * 72 * 2);
        const unsigned ob = (unsigned)(buf * 64 * 72 * 2);
        async_ld_b128(lA0 + oa,      ga0);
        async_ld_b128(lA0 + oa + 64, ga0 + 64);
        async_ld_b128(lA1 + oa,      ga1);
        async_ld_b128(lA1 + oa + 64, ga1 + 64);
        async_ld_b128(lB + ob,       gw);
        async_ld_b128(lB + ob + 64,  gw + 64);
        ga0 += 128; ga1 += 128; gw += 128;
        if (++ktc == 4) {            // k9 boundary (4 x 64 = 256 = full IC)
            ktc = 0;
            gw += wD;
            if (k3 == 2) { k3 = 0; ga0 += aD; ga1 += aD; }
            else { ++k3; }
        }
    };

    // ---- wave tile ----
    const int wm = (wv & 3) * 32;
    const int wn = (wv >> 2) * 32;
    v8f acc[2][2];
    for (int a = 0; a < 2; ++a)
        for (int b2 = 0; b2 < 2; ++b2)
            for (int i = 0; i < 8; ++i) acc[a][b2][i] = 0.f;

    // fragment read pointers (ISA 7.12.2 layouts), buffer 0
    const int row  = lane & 15;
    const int kb8  = (lane >> 4) * 8;
    const _Float16* fa0p = &As[(wm + row) * 72 + kb8];
    const _Float16* fa1p = &As[(wm + 16 + row) * 72 + kb8];
    const _Float16* fb0p = &Bs[(wn + row) * 72 + kb8];
    const _Float16* fb1p = &Bs[(wn + 16 + row) * 72 + kb8];

    auto compute = [&](int buf) {
        const int oa = buf * 128 * 72;
        const int ob = buf * 64 * 72;
        for (int c = 0; c < 2; ++c) {         // two 32-K chunks per stage
            const int ca = oa + c * 32;
            const int cb = ob + c * 32;
            v16h fa0 = cat16(*(const v8h*)(fa0p + ca), *(const v8h*)(fa0p + ca + 16));
            v16h fa1 = cat16(*(const v8h*)(fa1p + ca), *(const v8h*)(fa1p + ca + 16));
            v16h fb0 = cat16(*(const v8h*)(fb0p + cb), *(const v8h*)(fb0p + cb + 16));
            v16h fb1 = cat16(*(const v8h*)(fb1p + cb), *(const v8h*)(fb1p + cb + 16));
            acc[0][0] = __builtin_amdgcn_wmma_f32_16x16x32_f16(false, fa0, false, fb0, (short)0, acc[0][0], false, false);
            acc[0][1] = __builtin_amdgcn_wmma_f32_16x16x32_f16(false, fa0, false, fb1, (short)0, acc[0][1], false, false);
            acc[1][0] = __builtin_amdgcn_wmma_f32_16x16x32_f16(false, fa1, false, fb0, (short)0, acc[1][0], false, false);
            acc[1][1] = __builtin_amdgcn_wmma_f32_16x16x32_f16(false, fa1, false, fb1, (short)0, acc[1][1], false, false);
        }
    };

    // ---- software-pipelined main loop: 36 stages, double-buffered ----
    issue(0);
    s_wait_async0();
    __syncthreads();
    for (int s = 0; s < 36; s += 2) {
        issue(1);                       // stage s+1 (s+1 <= 35 always)
        compute(0);
        s_wait_async0();
        __syncthreads();
        if (s + 2 < 36) issue(0);       // stage s+2
        compute(1);
        s_wait_async0();
        __syncthreads();
    }

    // ---- epilogue: D elem r of lane l -> M = r + 8*(l>=16), N = l&15 ----
    const int nloc  = lane & 15;
    const int mb    = m0 + wm + ((lane >> 4) * 8);
    const float sc  = (mode == 2) ? scale[0] : 1.f;
    for (int ai = 0; ai < 2; ++ai) {
        for (int r = 0; r < 8; ++r) {
            const int mi = mb + ai * 16 + r;
            if (mi >= M) continue;
            const int bb = mi / HW;
            const int pp = mi - bb * HW;
            const int yy = pp / W;
            const int xx = pp - yy * W;
            const int oc0 = n0 + wn + nloc;
            const int oc1 = oc0 + 16;
            if (mode == 0) {
                const long long pidx =
                    ((long long)(bb * HW2 + (yy + 1) * W2 + (xx + 1))) * 256;
                out16[pidx + oc0] = (_Float16)(acc[ai][0][r] + bias[oc0]);
                out16[pidx + oc1] = (_Float16)(acc[ai][1][r] + bias[oc1]);
            } else {
                const long long base = (long long)bb * TOT + out_off + pp;
                if (oc0 < OC) {
                    float v0 = acc[ai][0][r] + bias[oc0];
                    if (mode == 2) v0 = expf(sc * v0);
                    out32[base + (long long)(ocbase + oc0) * HW] = v0;
                }
                if (oc1 < OC) {
                    float v1 = acc[ai][1][r] + bias[oc1];
                    if (mode == 2) v1 = expf(sc * v1);
                    out32[base + (long long)(ocbase + oc1) * HW] = v1;
                }
            }
        }
    }
}

// GroupNorm statistics over padded f16 NHWC: one WG per (batch, group).
__global__ __launch_bounds__(256) void gn_stats(
    const _Float16* __restrict__ xin, float* __restrict__ stats,
    int W, int HW, int W2, int HW2)
{
    __shared__ float ssum[256];
    __shared__ float ssq[256];
    const int t  = threadIdx.x;
    const int bg = blockIdx.x;     // b*32 + g
    const int g  = bg & 31;
    const int b  = bg >> 5;
    float s = 0.f, q = 0.f;
    for (int p = t; p < HW; p += 256) {
        const int y = p / W, x = p - y * W;
        const v8h v = *(const v8h*)(
            xin + ((long long)(b * HW2 + (y + 1) * W2 + (x + 1))) * 256 + g * 8);
        for (int j = 0; j < 8; ++j) { float f = (float)v[j]; s += f; q += f * f; }
    }
    ssum[t] = s; ssq[t] = q;
    __syncthreads();
    for (int st = 128; st > 0; st >>= 1) {
        if (t < st) { ssum[t] += ssum[t + st]; ssq[t] += ssq[t + st]; }
        __syncthreads();
    }
    if (t == 0) {
        const float inv = 1.f / (8.f * (float)HW);
        const float m   = ssum[0] * inv;
        const float var = ssq[0] * inv - m * m;
        stats[2 * bg]     = m;
        stats[2 * bg + 1] = rsqrtf(var + FCOS_EPS);
    }
}

// GroupNorm apply + ReLU on padded layout (8 channels per thread).
__global__ __launch_bounds__(256) void gn_apply(
    const _Float16* __restrict__ xin, _Float16* __restrict__ xout,
    const float* __restrict__ stats, const float* __restrict__ gamma,
    const float* __restrict__ beta, int W, int HW, int W2, int HW2, int total)
{
    const int idx = blockIdx.x * 256 + threadIdx.x;
    if (idx >= total) return;
    const int g = idx & 31;
    const int m = idx >> 5;
    const int b = m / HW;
    const int p = m - b * HW;
    const int y = p / W, x = p - y * W;
    const long long pidx = ((long long)(b * HW2 + (y + 1) * W2 + (x + 1))) * 256 + g * 8;
    const float mean = stats[2 * (b * 32 + g)];
    const float rstd = stats[2 * (b * 32 + g) + 1];
    v8h v = *(const v8h*)(xin + pidx);
    v8h o;
    for (int j = 0; j < 8; ++j) {
        const int c = g * 8 + j;
        float f = ((float)v[j] - mean) * rstd * gamma[c] + beta[c];
        o[j] = (_Float16)(f > 0.f ? f : 0.f);
    }
    *(v8h*)(xout + pidx) = o;
}

// NCHW f32 -> padded NHWC f16 (C = 256).
__global__ void nchw2nhwc(const float* __restrict__ in, _Float16* __restrict__ out,
                          int W, int HW, int W2, int HW2, long long total)
{
    const long long idx = (long long)blockIdx.x * 256 + threadIdx.x;
    if (idx >= total) return;
    const long long chw = 256LL * HW;
    const int b = (int)(idx / chw);
    const long long r = idx - (long long)b * chw;
    const int c = (int)(r / HW);
    const int p = (int)(r - (long long)c * HW);
    const int y = p / W, x = p - y * W;
    out[((long long)(b * HW2 + (y + 1) * W2 + (x + 1))) * 256 + c] = (_Float16)in[idx];
}

// [OC][256][3][3] f32 -> [9][OCpad][256] f16 (zero-padded out channels).
__global__ void wt_transform(const float* __restrict__ w, _Float16* __restrict__ wt,
                             int OC, int OCpad)
{
    const int idx = blockIdx.x * 256 + threadIdx.x;
    const int tot = 9 * OCpad * 256;
    if (idx >= tot) return;
    const int k9 = idx / (OCpad * 256);
    const int r  = idx - k9 * OCpad * 256;
    const int oc = r >> 8;
    const int ic = r & 255;
    float v = 0.f;
    if (oc < OC) v = w[((long long)(oc * 256 + ic)) * 9 + k9];
    wt[idx] = (_Float16)v;
}

__global__ void zero_f16(_Float16* __restrict__ p, long long n8)
{
    const long long i = (long long)blockIdx.x * 256 + threadIdx.x;
    if (i < n8) *(v8h*)(p + i * 8) = v8h_zero();
}

// FCOS grid locations for one level.
__global__ void locations_kern(float* __restrict__ dst, int H, int W, int s)
{
    const int idx = blockIdx.x * 256 + threadIdx.x;
    if (idx >= H * W) return;
    const int y = idx / W, x = idx - y * W;
    dst[2 * idx]     = (float)(x * s + s / 2);
    dst[2 * idx + 1] = (float)(y * s + s / 2);
}

extern "C" void kernel_launch(void* const* d_in, const int* in_sizes, int n_in,
                              void* d_out, int out_size, void* d_ws, size_t ws_size,
                              hipStream_t stream)
{
    (void)in_sizes; (void)n_in; (void)out_size; (void)ws_size;
    const float* feats[5] = {
        (const float*)d_in[0], (const float*)d_in[1], (const float*)d_in[2],
        (const float*)d_in[3], (const float*)d_in[4]};
    const float* cls_w    = (const float*)d_in[5];
    const float* cls_b    = (const float*)d_in[6];
    const float* cls_gn_g = (const float*)d_in[7];
    const float* cls_gn_b = (const float*)d_in[8];
    const float* box_w    = (const float*)d_in[9];
    const float* box_b    = (const float*)d_in[10];
    const float* box_gn_g = (const float*)d_in[11];
    const float* box_gn_b = (const float*)d_in[12];
    const float* logits_w = (const float*)d_in[13];
    const float* logits_b = (const float*)d_in[14];
    const float* ctr_w    = (const float*)d_in[15];
    const float* ctr_b    = (const float*)d_in[16];
    const float* reg_w    = (const float*)d_in[17];
    const float* reg_b    = (const float*)d_in[18];
    const float* scales   = (const float*)d_in[19];
    float* out = (float*)d_out;

    static const int Hs[5] = {100, 50, 25, 13, 7};
    static const int Ws_[5] = {152, 76, 38, 19, 10};
    static const int Ss[5] = {8, 16, 32, 64, 128};
    const int Bsz = 8;
    long long cumHW[6]; cumHW[0] = 0;
    for (int l = 0; l < 5; ++l) cumHW[l + 1] = cumHW[l] + (long long)Hs[l] * Ws_[l];
    const long long TOT = 85 * cumHW[5];

    // workspace carve-up (padded activation buffers)
    char* w = (char*)d_ws;
    const long long CAPP = (long long)Bsz * (Hs[0] + 2) * (Ws_[0] + 2) * 256;  // elems
    _Float16* F    = (_Float16*)w; w += CAPP * 2;
    _Float16* T    = (_Float16*)w; w += CAPP * 2;
    _Float16* Abuf = (_Float16*)w; w += CAPP * 2;
    _Float16* Bbuf = (_Float16*)w; w += CAPP * 2;
    _Float16* wt_cls = (_Float16*)w; w += 4LL * 9 * 256 * 256 * 2;
    _Float16* wt_box = (_Float16*)w; w += 4LL * 9 * 256 * 256 * 2;
    _Float16* wt_log = (_Float16*)w; w += 9LL * 128 * 256 * 2;
    _Float16* wt_ctr = (_Float16*)w; w += 9LL * 64 * 256 * 2;
    _Float16* wt_reg = (_Float16*)w; w += 9LL * 64 * 256 * 2;
    float* stats = (float*)w; w += 512 * 4;

    // weight transforms
    for (int i = 0; i < 4; ++i) {
        const int tot = 9 * 256 * 256;
        wt_transform<<<(tot + 255) / 256, 256, 0, stream>>>(
            cls_w + (long long)i * 256 * 256 * 9,
            wt_cls + (long long)i * 9 * 256 * 256, 256, 256);
        wt_transform<<<(tot + 255) / 256, 256, 0, stream>>>(
            box_w + (long long)i * 256 * 256 * 9,
            wt_box + (long long)i * 9 * 256 * 256, 256, 256);
    }
    {
        int tot = 9 * 128 * 256;
        wt_transform<<<(tot + 255) / 256, 256, 0, stream>>>(logits_w, wt_log, 80, 128);
        tot = 9 * 64 * 256;
        wt_transform<<<(tot + 255) / 256, 256, 0, stream>>>(ctr_w, wt_ctr, 1, 64);
        wt_transform<<<(tot + 255) / 256, 256, 0, stream>>>(reg_w, wt_reg, 4, 64);
    }

    for (int l = 0; l < 5; ++l) {
        const int H = Hs[l], W = Ws_[l], HW = H * W, M = Bsz * HW;
        const int W2 = W + 2, HW2 = (H + 2) * W2;
        const long long off = 85 * cumHW[l];

        // zero halo-padded conv-input buffers for this level's geometry
        {
            const long long n8 = ((long long)Bsz * HW2 * 256) / 8;
            const unsigned g = (unsigned)((n8 + 255) / 256);
            zero_f16<<<g, 256, 0, stream>>>(F, n8);
            zero_f16<<<g, 256, 0, stream>>>(Abuf, n8);
            zero_f16<<<g, 256, 0, stream>>>(Bbuf, n8);
        }
        const long long tot = (long long)Bsz * 256 * HW;
        nchw2nhwc<<<(unsigned)((tot + 255) / 256), 256, 0, stream>>>(
            feats[l], F, W, HW, W2, HW2, tot);

        const int nM = (M + 127) / 128;
        dim3 cg(nM, 4);
        _Float16* dsts[4] = {Abuf, Bbuf, Abuf, Bbuf};

        // cls tower
        const _Float16* src = F;
        for (int i = 0; i < 4; ++i) {
            conv3x3_wmma<<<cg, 256, 0, stream>>>(
                src, wt_cls + (long long)i * 9 * 256 * 256, cls_b + i * 256,
                T, nullptr, nullptr, Bsz, W, HW, W2, HW2, 256, 256, 0, 0LL, 0LL, 0);
            gn_stats<<<256, 256, 0, stream>>>(T, stats, W, HW, W2, HW2);
            gn_apply<<<(M * 32 + 255) / 256, 256, 0, stream>>>(
                T, dsts[i], stats, cls_gn_g + i * 256, cls_gn_b + i * 256,
                W, HW, W2, HW2, M * 32);
            src = dsts[i];
        }
        // heads from cls tower (final in Bbuf)
        conv3x3_wmma<<<dim3(nM, 2), 256, 0, stream>>>(
            Bbuf, wt_log, logits_b, nullptr, out, nullptr,
            Bsz, W, HW, W2, HW2, 80, 128, 1, off, TOT, 0);
        conv3x3_wmma<<<dim3(nM, 1), 256, 0, stream>>>(
            Bbuf, wt_ctr, ctr_b, nullptr, out, nullptr,
            Bsz, W, HW, W2, HW2, 1, 64, 1, off, TOT, 84);

        // box tower
        src = F;
        for (int i = 0; i < 4; ++i) {
            conv3x3_wmma<<<cg, 256, 0, stream>>>(
                src, wt_box + (long long)i * 9 * 256 * 256, box_b + i * 256,
                T, nullptr, nullptr, Bsz, W, HW, W2, HW2, 256, 256, 0, 0LL, 0LL, 0);
            gn_stats<<<256, 256, 0, stream>>>(T, stats, W, HW, W2, HW2);
            gn_apply<<<(M * 32 + 255) / 256, 256, 0, stream>>>(
                T, dsts[i], stats, box_gn_g + i * 256, box_gn_b + i * 256,
                W, HW, W2, HW2, M * 32);
            src = dsts[i];
        }
        conv3x3_wmma<<<dim3(nM, 1), 256, 0, stream>>>(
            Bbuf, wt_reg, reg_b, nullptr, out, scales + l,
            Bsz, W, HW, W2, HW2, 4, 64, 2, off, TOT, 80);

        // locations
        locations_kern<<<(HW + 255) / 256, 256, 0, stream>>>(
            out + 8 * TOT + 2 * cumHW[l], H, W, Ss[l]);
    }
}
